// GNNModel_10033043603774
// MI455X (gfx1250) — compile-verified, matrix operand order checked by reading, not run on hardware
//
#include <hip/hip_runtime.h>

typedef _Float16 half_t;
typedef __attribute__((ext_vector_type(16))) _Float16 v16h;
typedef __attribute__((ext_vector_type(8)))  _Float16 v8h;
typedef __attribute__((ext_vector_type(8)))  float    v8f;

#define DEVINL __device__ __forceinline__

// ---------------------------------------------------------------------------
// WMMA helpers (gfx1250: V_WMMA_F32_16X16X32_F16, wave32)
// ---------------------------------------------------------------------------
DEVINL v8f wmma_f16(v16h a, v16h b, v8f c) {
  return __builtin_amdgcn_wmma_f32_16x16x32_f16(false, a, false, b, (short)0, c,
                                                false, false);
}

// Build one 16-halfword fragment for the documented 16-bit A/B layout:
// lane holds row (M for A, N for B) = lane&15; half = lane>>4 selects K groups.
// K(e,half) = e + 8*half          (e in 0..7)
//           = e + 8 + 8*half      (e in 8..15)
// => two contiguous 16B loads at k0+8*half and k0+16+8*half.
DEVINL v16h load_frag(const half_t* rowbase, int k0) {
  const int hh = (threadIdx.x >> 4) & 1;
  const v8h p0 = *(const v8h*)(rowbase + k0 + 8 * hh);
  const v8h p1 = *(const v8h*)(rowbase + k0 + 16 + 8 * hh);
  v16h f;
#pragma unroll
  for (int i = 0; i < 8; ++i) { f[i] = p0[i]; f[8 + i] = p1[i]; }
  return f;
}

template <int NT>
DEVINL void zacc(v8f* a) {
  v8f z = {0.f, 0.f, 0.f, 0.f, 0.f, 0.f, 0.f, 0.f};
#pragma unroll
  for (int i = 0; i < NT; ++i) a[i] = z;
}

// D = A(16xK) * W^T (+acc).  Atile: LDS tile, row stride lda halves.
// W: row-major [OUT, ldk] f16 (B[k][n] = W[n][k]).
template <int KS, int NT>
DEVINL void gemm16(const half_t* Atile, int lda, const half_t* W, int ldk,
                   v8f* acc) {
  const int nl = threadIdx.x & 15;
  const half_t* arow = Atile + (threadIdx.x & 15) * lda;
#pragma unroll
  for (int ks = 0; ks < KS; ++ks) {
    v16h a = load_frag(arow, ks * 32);
#pragma unroll
    for (int nt = 0; nt < NT; ++nt) {
      v16h b = load_frag(W + (nt * 16 + nl) * ldk, ks * 32);
      acc[nt] = wmma_f16(a, b, acc[nt]);
    }
  }
}

// Store D tile (+bias, activation) as f16 into LDS tile.
// ACT: 0 none, 1 relu, 2 leaky(0.01), 3 elu
template <int NT, int ACT>
DEVINL void store_tile(const v8f* acc, const float* bias, half_t* Otile,
                       int ldo) {
  const int nl = threadIdx.x & 15, hh = (threadIdx.x >> 4) & 1;
#pragma unroll
  for (int nt = 0; nt < NT; ++nt) {
    float bv = bias[nt * 16 + nl];
#pragma unroll
    for (int r = 0; r < 8; ++r) {
      float v = acc[nt][r] + bv;
      if (ACT == 1) v = fmaxf(v, 0.f);
      if (ACT == 2) v = (v > 0.f) ? v : 0.01f * v;
      if (ACT == 3) v = (v > 0.f) ? v : (expf(v) - 1.f);
      Otile[(r + 8 * hh) * ldo + nt * 16 + nl] = (half_t)v;
    }
  }
}

DEVINL float wredsum(float v) {
#pragma unroll
  for (int m = 16; m > 0; m >>= 1) v += __shfl_xor(v, m, 32);
  return v;
}

// ---------------------------------------------------------------------------
// f32 -> f16 weight conversion
// ---------------------------------------------------------------------------
__global__ void k_f32_to_f16(const float* __restrict__ s, half_t* __restrict__ d,
                             int n) {
  int i = blockIdx.x * blockDim.x + threadIdx.x;
  if (i < n) d[i] = (half_t)s[i];
}

// ---------------------------------------------------------------------------
// Node preprocess: invalid-replace + LayerNorm(64) -> f16  (one wave / node)
// ---------------------------------------------------------------------------
__global__ void k_node_ln(const float* __restrict__ xin,
                          const float* __restrict__ dummy,
                          const float* __restrict__ g,
                          const float* __restrict__ b,
                          half_t* __restrict__ xf16, int N) {
  const int wid = threadIdx.x >> 5, lane = threadIdx.x & 31;
  int node = blockIdx.x * (blockDim.x >> 5) + wid;
  int nc = node < N ? node : N - 1;
  const float* xr = xin + (long)nc * 64;
  float v0 = xr[2 * lane], v1 = xr[2 * lane + 1];
  float x0 = __shfl(v0, 0, 32);
  if (x0 == -999.0f) { v0 = dummy[2 * lane]; v1 = dummy[2 * lane + 1]; }
  float mean = wredsum(v0 + v1) * (1.f / 64.f);
  float d0 = v0 - mean, d1 = v1 - mean;
  float var = wredsum(d0 * d0 + d1 * d1) * (1.f / 64.f);
  float rstd = rsqrtf(var + 1e-6f);
  if (node < N) {
    xf16[(long)nc * 64 + 2 * lane] = (half_t)(d0 * rstd * g[2 * lane] + b[2 * lane]);
    xf16[(long)nc * 64 + 2 * lane + 1] =
        (half_t)(d1 * rstd * g[2 * lane + 1] + b[2 * lane + 1]);
  }
}

// ---------------------------------------------------------------------------
// Edge encoder + classifier.  One wave = 16 edges.
// LN(ea) -> relu(W1) -> relu(W2) -> W3, scaled by sigmoid classifier weight.
// Emits e_enc f16 [E,128], fp32 atomics into e_sum [N,128] and cnt [N].
// ---------------------------------------------------------------------------
__global__ void k_edge_encoder(
    const float* __restrict__ ea, const int* __restrict__ dstI,
    const float* __restrict__ lng, const float* __restrict__ lnb,
    const half_t* __restrict__ W1, const float* __restrict__ b1,
    const half_t* __restrict__ W2, const float* __restrict__ b2,
    const half_t* __restrict__ W3, const float* __restrict__ b3,
    const half_t* __restrict__ cW1, const float* __restrict__ cb1,
    const float* __restrict__ cW2, const float* __restrict__ cb2,
    half_t* __restrict__ eenc, float* __restrict__ esum,
    float* __restrict__ cnt, int E) {
  __shared__ __align__(16) half_t sA[4][16 * 256];
  __shared__ __align__(16) half_t sB[4][16 * 128];
  __shared__ float sW[4][16];
  __shared__ int sD[4][16];
  const int wid = threadIdx.x >> 5, lane = threadIdx.x & 31;
  const int nl = lane & 15, hh = lane >> 4;
  half_t* A = sA[wid];
  half_t* Bb = sB[wid];
  float* wv = sW[wid];
  int* dv = sD[wid];
  const int e0 = (blockIdx.x * (blockDim.x >> 5) + wid) * 16;

  // ---- LayerNorm(ea) into A[16][32] (ld 256); 2 lanes per edge row ----
  {
    const int m = nl;
    int edge = e0 + m;
    int ec = edge < E ? edge : E - 1;
    const float* er = ea + (long)ec * 32 + hh * 16;
    float t[16];
    float s = 0.f;
#pragma unroll
    for (int i = 0; i < 16; ++i) { t[i] = er[i]; s += t[i]; }
    s += __shfl_xor(s, 16, 32);
    float mean = s * (1.f / 32.f);
    float vs = 0.f;
#pragma unroll
    for (int i = 0; i < 16; ++i) { float d = t[i] - mean; vs += d * d; }
    vs += __shfl_xor(vs, 16, 32);
    float rstd = rsqrtf(vs * (1.f / 32.f) + 1e-6f);
#pragma unroll
    for (int i = 0; i < 16; ++i) {
      int c = hh * 16 + i;
      A[m * 256 + c] = (half_t)((t[i] - mean) * rstd * lng[c] + lnb[c]);
    }
    if (hh == 0) dv[m] = dstI[ec];
  }

  // ---- classifier: h = relu(LNea? no: raw ea)  ----
  // Reference uses raw ea for the classifier; recompute raw tile into Bb.
  {
    const int m = nl;
    int edge = e0 + m;
    int ec = edge < E ? edge : E - 1;
    const float* er = ea + (long)ec * 32 + hh * 16;
#pragma unroll
    for (int i = 0; i < 16; ++i) Bb[m * 128 + hh * 16 + i] = (half_t)er[i];
  }
  {
    v8f acc[2];
    zacc<2>(acc);
    gemm16<1, 2>(Bb, 128, cW1, 32, acc);
    store_tile<2, 1>(acc, cb1, Bb, 128);  // relu, cols 0..31 of Bb rows
    if (hh == 0) {
      float z = cb2[0];
#pragma unroll
      for (int k = 0; k < 32; ++k) z += (float)Bb[nl * 128 + k] * cW2[k];
      wv[nl] = 1.f / (1.f + expf(-z));
    }
  }

  // ---- L1: [16,32] x (32->128) relu -> Bb[16][128] ----
  {
    v8f acc[8];
    zacc<8>(acc);
    gemm16<1, 8>(A, 256, W1, 32, acc);
    store_tile<8, 1>(acc, b1, Bb, 128);
  }
  // ---- L2: [16,128] x (128->256) relu -> A[16][256]  (two 128-col halves) ----
  {
    v8f acc[8];
    zacc<8>(acc);
    gemm16<4, 8>(Bb, 128, W2, 128, acc);
    store_tile<8, 1>(acc, b2, A, 256);
    zacc<8>(acc);
    gemm16<4, 8>(Bb, 128, W2 + 128 * 128, 128, acc);
    store_tile<8, 1>(acc, b2 + 128, A + 128, 256);
  }
  // ---- L3: [16,256] x (256->128), scale by w, emit ----
  {
    v8f acc[8];
    zacc<8>(acc);
    gemm16<8, 8>(A, 256, W3, 256, acc);
#pragma unroll
    for (int nt = 0; nt < 8; ++nt) {
      float bv = b3[nt * 16 + nl];
#pragma unroll
      for (int r = 0; r < 8; ++r) {
        int m = r + 8 * hh;
        int edge = e0 + m;
        if (edge < E) {
          int col = nt * 16 + nl;
          float v = (acc[nt][r] + bv) * wv[m];
          eenc[(long)edge * 128 + col] = (half_t)v;
          atomicAdd(&esum[(long)dv[m] * 128 + col], v);
        }
      }
    }
  }
  if (hh == 0 && (e0 + nl) < E) atomicAdd(&cnt[dv[nl]], 1.f);
}

// ---------------------------------------------------------------------------
// EdgeConv message MLP (OD = 64 for conv1, 128 for conv2).
// msg = cat[x[dst](64), x[src](64), e_enc(128)] -> 256 -> OD -> OD -> OD,
// fp32 atomic accumulate into osum[dst].
// ---------------------------------------------------------------------------
template <int OD>
__global__ void k_edge_conv(const half_t* __restrict__ xh,
                            const half_t* __restrict__ eenc,
                            const int* __restrict__ srcI,
                            const int* __restrict__ dstI,
                            const half_t* __restrict__ W1,
                            const float* __restrict__ b1,
                            const half_t* __restrict__ W2,
                            const float* __restrict__ b2,
                            const half_t* __restrict__ W3,
                            const float* __restrict__ b3,
                            float* __restrict__ osum, int E) {
  constexpr int NT = OD / 16;
  constexpr int KS2 = OD / 32;
  __shared__ __align__(16) half_t sA[4][16 * 256];
  __shared__ __align__(16) half_t sB[4][16 * OD];
  __shared__ int sD[4][16];
  const int wid = threadIdx.x >> 5, lane = threadIdx.x & 31;
  const int nl = lane & 15, hh = lane >> 4;
  half_t* A = sA[wid];
  half_t* Bb = sB[wid];
  int* dv = sD[wid];
  const int e0 = (blockIdx.x * (blockDim.x >> 5) + wid) * 16;

  // ---- stage message tile: each lane copies one 16B chunk per edge ----
#pragma unroll 1
  for (int m = 0; m < 16; ++m) {
    int edge = e0 + m;
    int ec = edge < E ? edge : E - 1;
    int s = srcI[ec], d = dstI[ec];
    const half_t* p;
    if (lane < 8)       p = xh + (long)d * 64 + 8 * lane;
    else if (lane < 16) p = xh + (long)s * 64 + 8 * (lane - 8);
    else                p = eenc + (long)ec * 128 + 8 * (lane - 16);
    *(v8h*)(A + m * 256 + 8 * lane) = *(const v8h*)p;
    if (lane == 0) dv[m] = d;
  }

  // ---- L1: 256 -> OD relu ----
  {
    v8f acc[NT];
    zacc<NT>(acc);
    gemm16<8, NT>(A, 256, W1, 256, acc);
    store_tile<NT, 1>(acc, b1, Bb, OD);
  }
  // ---- L2: OD -> OD relu (in place: LDS ops are wave-in-order) ----
  {
    v8f acc[NT];
    zacc<NT>(acc);
    gemm16<KS2, NT>(Bb, OD, W2, OD, acc);
    store_tile<NT, 1>(acc, b2, Bb, OD);
  }
  // ---- L3: OD -> OD, atomic scatter to dst ----
  {
    v8f acc[NT];
    zacc<NT>(acc);
    gemm16<KS2, NT>(Bb, OD, W3, OD, acc);
#pragma unroll
    for (int nt = 0; nt < NT; ++nt) {
      float bv = b3[nt * 16 + nl];
#pragma unroll
      for (int r = 0; r < 8; ++r) {
        int m = r + 8 * hh;
        int edge = e0 + m;
        if (edge < E) {
          int col = nt * 16 + nl;
          atomicAdd(&osum[(long)dv[m] * OD + col], acc[nt][r] + bv);
        }
      }
    }
  }
}

// ---------------------------------------------------------------------------
// x1 finalize: mean-div, LN(64), leaky_relu -> f16     (one wave / node)
// ---------------------------------------------------------------------------
__global__ void k_x1_fin(const float* __restrict__ s, const float* __restrict__ cnt,
                         const float* __restrict__ g, const float* __restrict__ b,
                         half_t* __restrict__ x1f, int N) {
  const int wid = threadIdx.x >> 5, lane = threadIdx.x & 31;
  int node = blockIdx.x * (blockDim.x >> 5) + wid;
  int nc = node < N ? node : N - 1;
  float deg = fmaxf(cnt[nc], 1.f);
  float v0 = s[(long)nc * 64 + 2 * lane] / deg;
  float v1 = s[(long)nc * 64 + 2 * lane + 1] / deg;
  float mean = wredsum(v0 + v1) * (1.f / 64.f);
  float d0 = v0 - mean, d1 = v1 - mean;
  float var = wredsum(d0 * d0 + d1 * d1) * (1.f / 64.f);
  float rstd = rsqrtf(var + 1e-6f);
  float y0 = d0 * rstd * g[2 * lane] + b[2 * lane];
  float y1 = d1 * rstd * g[2 * lane + 1] + b[2 * lane + 1];
  y0 = y0 > 0.f ? y0 : 0.01f * y0;
  y1 = y1 > 0.f ? y1 : 0.01f * y1;
  if (node < N) {
    x1f[(long)nc * 64 + 2 * lane] = (half_t)y0;
    x1f[(long)nc * 64 + 2 * lane + 1] = (half_t)y1;
  }
}

// ---------------------------------------------------------------------------
// x2 finalize: mean-div, LN(128), relu -> f16 ; e_mean -> f16 + out[:,128:256]
// ---------------------------------------------------------------------------
__global__ void k_x2_fin(const float* __restrict__ x2s, const float* __restrict__ es,
                         const float* __restrict__ cnt, const float* __restrict__ g,
                         const float* __restrict__ b, half_t* __restrict__ x2f,
                         half_t* __restrict__ emf, float* __restrict__ out, int N) {
  const int wid = threadIdx.x >> 5, lane = threadIdx.x & 31;
  int node = blockIdx.x * (blockDim.x >> 5) + wid;
  int nc = node < N ? node : N - 1;
  float deg = fmaxf(cnt[nc], 1.f);
  float t[4];
  float s = 0.f;
#pragma unroll
  for (int i = 0; i < 4; ++i) {
    t[i] = x2s[(long)nc * 128 + 4 * lane + i] / deg;
    s += t[i];
  }
  float mean = wredsum(s) * (1.f / 128.f);
  float vs = 0.f;
#pragma unroll
  for (int i = 0; i < 4; ++i) { float d = t[i] - mean; vs += d * d; }
  float var = wredsum(vs) * (1.f / 128.f);
  float rstd = rsqrtf(var + 1e-6f);
#pragma unroll
  for (int i = 0; i < 4; ++i) {
    int c = 4 * lane + i;
    float y = (t[i] - mean) * rstd * g[c] + b[c];
    y = fmaxf(y, 0.f);
    float em = es[(long)nc * 128 + c] / deg;
    if (node < N) {
      x2f[(long)nc * 128 + c] = (half_t)y;
      emf[(long)nc * 128 + c] = (half_t)em;
      out[(long)nc * 256 + 128 + c] = em;
    }
  }
}

// ---------------------------------------------------------------------------
// Node head: skip/gate/xf + np MLP.  One wave = 16 nodes.
// ---------------------------------------------------------------------------
__global__ void k_node_final(
    const half_t* __restrict__ xh, const half_t* __restrict__ x2f,
    const half_t* __restrict__ emf, const half_t* __restrict__ skW,
    const float* __restrict__ skb, const half_t* __restrict__ gW,
    const float* __restrict__ gb, const half_t* __restrict__ nW1,
    const float* __restrict__ nb1, const half_t* __restrict__ nW2,
    const float* __restrict__ nb2, const float* __restrict__ nW3,
    const float* __restrict__ nb3, float* __restrict__ out, int N) {
  __shared__ __align__(16) half_t sA[4][16 * 256];
  __shared__ __align__(16) half_t sC[4][16 * 128];
  const int wid = threadIdx.x >> 5, lane = threadIdx.x & 31;
  const int nl = lane & 15, hh = lane >> 4;
  half_t* A = sA[wid];
  half_t* C = sC[wid];
  const int n0 = (blockIdx.x * (blockDim.x >> 5) + wid) * 16;

  // stage x (cols 0..63) and e_mean (cols 128..255)
#pragma unroll 1
  for (int m = 0; m < 16; ++m) {
    int node = n0 + m;
    int nc = node < N ? node : N - 1;
    if (lane < 8)
      *(v8h*)(A + m * 256 + 8 * lane) = *(const v8h*)(xh + (long)nc * 64 + 8 * lane);
    else if (lane >= 16)
      *(v8h*)(A + m * 256 + 128 + 8 * (lane - 16)) =
          *(const v8h*)(emf + (long)nc * 128 + 8 * (lane - 16));
  }

  // skip = x @ skW^T + skb  -> f16 at A cols 64..191
  {
    v8f acc[8];
    zacc<8>(acc);
    gemm16<2, 8>(A, 256, skW, 64, acc);
    store_tile<8, 0>(acc, skb, A + 64, 256);
  }
  // gate = sigmoid(skip @ gW^T + gb);  xf = g*skip + (1-g)*x2  -> C + out
  {
    v8f acc[8];
    zacc<8>(acc);
    gemm16<4, 8>(A + 64, 256, gW, 128, acc);
#pragma unroll
    for (int nt = 0; nt < 8; ++nt) {
      float bv = gb[nt * 16 + nl];
#pragma unroll
      for (int r = 0; r < 8; ++r) {
        int m = r + 8 * hh;
        int node = n0 + m;
        int nc = node < N ? node : N - 1;
        int col = nt * 16 + nl;
        float gs = 1.f / (1.f + expf(-(acc[nt][r] + bv)));
        float sk = (float)A[m * 256 + 64 + col];
        float x2v = (float)x2f[(long)nc * 128 + col];
        float xv = gs * sk + (1.f - gs) * x2v;
        C[m * 128 + col] = (half_t)xv;
        if (node < N) out[(long)node * 256 + col] = xv;
      }
    }
  }
  // np1: [16,256] (C: xf | A+128: e_mean) x (256->128), elu -> A cols 0..127
  {
    v8f acc[8];
    zacc<8>(acc);
    const half_t* ar0 = C + (threadIdx.x & 15) * 128;
    const half_t* ar1 = A + (threadIdx.x & 15) * 256 + 128;
#pragma unroll
    for (int ks = 0; ks < 8; ++ks) {
      v16h a = load_frag(ks < 4 ? ar0 : ar1, (ks & 3) * 32);
#pragma unroll
      for (int nt = 0; nt < 8; ++nt) {
        v16h b = load_frag(nW1 + (nt * 16 + nl) * 256, ks * 32);
        acc[nt] = wmma_f16(a, b, acc[nt]);
      }
    }
    store_tile<8, 3>(acc, nb1, A, 256);
  }
  // np2: 128 -> 64, elu -> C cols 0..63
  {
    v8f acc[4];
    zacc<4>(acc);
    gemm16<4, 4>(A, 256, nW2, 128, acc);
    store_tile<4, 3>(acc, nb2, C, 128);
  }
  // np3: 64 -> 1 (per-node dot, lanes 0..15)
  if (hh == 0) {
    int node = n0 + nl;
    float z = nb3[0];
#pragma unroll
    for (int k = 0; k < 64; ++k) z += (float)C[nl * 128 + k] * nW3[k];
    if (node < N) out[(long)N * 256 + node] = z;
  }
}

// ---------------------------------------------------------------------------
// Host launch
// ---------------------------------------------------------------------------
extern "C" void kernel_launch(void* const* d_in, const int* in_sizes, int n_in,
                              void* d_out, int out_size, void* d_ws,
                              size_t ws_size, hipStream_t stream) {
  const float* x_in = (const float*)d_in[0];
  const int* eidx = (const int*)d_in[1];
  const float* ea = (const float*)d_in[2];
  const float* p_dummy = (const float*)d_in[3];
  const float* bn0_g = (const float*)d_in[4];
  const float* bn0_b = (const float*)d_in[5];
  const float* skip_W = (const float*)d_in[6];
  const float* skip_b = (const float*)d_in[7];
  const float* enc_ln_g = (const float*)d_in[8];
  const float* enc_ln_b = (const float*)d_in[9];
  const float* enc_W1 = (const float*)d_in[10];
  const float* enc_b1 = (const float*)d_in[11];
  const float* enc_W2 = (const float*)d_in[12];
  const float* enc_b2 = (const float*)d_in[13];
  const float* enc_W3 = (const float*)d_in[14];
  const float* enc_b3 = (const float*)d_in[15];
  const float* cls_W1 = (const float*)d_in[16];
  const float* cls_b1 = (const float*)d_in[17];
  const float* cls_W2 = (const float*)d_in[18];
  const float* cls_b2 = (const float*)d_in[19];
  const float* c1_W1 = (const float*)d_in[20];
  const float* c1_b1 = (const float*)d_in[21];
  const float* c1_W2 = (const float*)d_in[22];
  const float* c1_b2 = (const float*)d_in[23];
  const float* c1_W3 = (const float*)d_in[24];
  const float* c1_b3 = (const float*)d_in[25];
  const float* c2_W1 = (const float*)d_in[26];
  const float* c2_b1 = (const float*)d_in[27];
  const float* c2_W2 = (const float*)d_in[28];
  const float* c2_b2 = (const float*)d_in[29];
  const float* c2_W3 = (const float*)d_in[30];
  const float* c2_b3 = (const float*)d_in[31];
  const float* bn1_g = (const float*)d_in[32];
  const float* bn1_b = (const float*)d_in[33];
  const float* bn2_g = (const float*)d_in[34];
  const float* bn2_b = (const float*)d_in[35];
  const float* gate_W = (const float*)d_in[36];
  const float* gate_b = (const float*)d_in[37];
  const float* np_W1 = (const float*)d_in[38];
  const float* np_b1 = (const float*)d_in[39];
  const float* np_W2 = (const float*)d_in[40];
  const float* np_b2 = (const float*)d_in[41];
  const float* np_W3 = (const float*)d_in[42];
  const float* np_b3 = (const float*)d_in[43];

  const int N = in_sizes[0] / 64;
  const int E = in_sizes[2] / 32;
  const int* srcI = eidx;
  const int* dstI = eidx + E;

  // ---- workspace layout ----
  char* w = (char*)d_ws;
  auto take = [&](size_t bytes) -> char* {
    char* p = w;
    w += (bytes + 15) & ~(size_t)15;
    return p;
  };
  float* e_sum = (float*)take((size_t)N * 128 * 4);
  float* cnt = (float*)take((size_t)N * 4);
  float* x1sum = (float*)take((size_t)N * 64 * 4);
  float* x2sum = (float*)take((size_t)N * 128 * 4);
  size_t accbytes = (size_t)(w - (char*)d_ws);
  half_t* xf16 = (half_t*)take((size_t)N * 64 * 2);
  half_t* x1f16 = (half_t*)take((size_t)N * 64 * 2);
  half_t* x2f16 = (half_t*)take((size_t)N * 128 * 2);
  half_t* emf16 = (half_t*)take((size_t)N * 128 * 2);
  half_t* eenc = (half_t*)take((size_t)E * 128 * 2);

  half_t* hEncW1 = (half_t*)take(128 * 32 * 2);
  half_t* hEncW2 = (half_t*)take(256 * 128 * 2);
  half_t* hEncW3 = (half_t*)take(128 * 256 * 2);
  half_t* hClsW1 = (half_t*)take(32 * 32 * 2);
  half_t* hC1W1 = (half_t*)take(64 * 256 * 2);
  half_t* hC1W2 = (half_t*)take(64 * 64 * 2);
  half_t* hC1W3 = (half_t*)take(64 * 64 * 2);
  half_t* hC2W1 = (half_t*)take(128 * 256 * 2);
  half_t* hC2W2 = (half_t*)take(128 * 128 * 2);
  half_t* hC2W3 = (half_t*)take(128 * 128 * 2);
  half_t* hSkW = (half_t*)take(128 * 64 * 2);
  half_t* hGW = (half_t*)take(128 * 128 * 2);
  half_t* hNpW1 = (half_t*)take(128 * 256 * 2);
  half_t* hNpW2 = (half_t*)take(64 * 128 * 2);

  auto cvt = [&](const float* s, half_t* d, int n) {
    k_f32_to_f16<<<(n + 255) / 256, 256, 0, stream>>>(s, d, n);
  };
  cvt(enc_W1, hEncW1, 128 * 32);
  cvt(enc_W2, hEncW2, 256 * 128);
  cvt(enc_W3, hEncW3, 128 * 256);
  cvt(cls_W1, hClsW1, 32 * 32);
  cvt(c1_W1, hC1W1, 64 * 256);
  cvt(c1_W2, hC1W2, 64 * 64);
  cvt(c1_W3, hC1W3, 64 * 64);
  cvt(c2_W1, hC2W1, 128 * 256);
  cvt(c2_W2, hC2W2, 128 * 128);
  cvt(c2_W3, hC2W3, 128 * 128);
  cvt(skip_W, hSkW, 128 * 64);
  cvt(gate_W, hGW, 128 * 128);
  cvt(np_W1, hNpW1, 128 * 256);
  cvt(np_W2, hNpW2, 64 * 128);

  hipMemsetAsync(d_ws, 0, accbytes, stream);

  const int nodeWaves = (N + 3) / 4;       // 1 wave/node, 4 waves/block
  const int edgeBlocks = (E + 63) / 64;    // 16 edges/wave, 4 waves/block
  const int tileBlocks = (N + 63) / 64;    // 16 nodes/wave, 4 waves/block

  k_node_ln<<<nodeWaves, 128, 0, stream>>>(x_in, p_dummy, bn0_g, bn0_b, xf16, N);

  k_edge_encoder<<<edgeBlocks, 128, 0, stream>>>(
      ea, dstI, enc_ln_g, enc_ln_b, hEncW1, enc_b1, hEncW2, enc_b2, hEncW3,
      enc_b3, hClsW1, cls_b1, cls_W2, cls_b2, eenc, e_sum, cnt, E);

  k_edge_conv<64><<<edgeBlocks, 128, 0, stream>>>(
      xf16, eenc, srcI, dstI, hC1W1, c1_b1, hC1W2, c1_b2, hC1W3, c1_b3, x1sum, E);

  k_x1_fin<<<nodeWaves, 128, 0, stream>>>(x1sum, cnt, bn1_g, bn1_b, x1f16, N);

  k_edge_conv<128><<<edgeBlocks, 128, 0, stream>>>(
      x1f16, eenc, srcI, dstI, hC2W1, c2_b1, hC2W2, c2_b2, hC2W3, c2_b3, x2sum, E);

  k_x2_fin<<<nodeWaves, 128, 0, stream>>>(x2sum, e_sum, cnt, bn2_g, bn2_b, x2f16,
                                          emf16, (float*)d_out, N);

  k_node_final<<<tileBlocks, 128, 0, stream>>>(
      xf16, x2f16, emf16, hSkW, skip_b, hGW, gate_b, hNpW1, np_b1, hNpW2, np_b2,
      np_W3, np_b3, (float*)d_out, N);

  (void)n_in;
  (void)out_size;
  (void)ws_size;
}
